// SweetNet_9809705305013
// MI455X (gfx1250) — matrix operand order repaired and checked by reading, not compiled, not run on hardware
//
#include <hip/hip_runtime.h>
#include <hip/hip_bf16.h>

// ---------------------------------------------------------------------------
// SweetNet forward for MI455X (gfx1250): bf16 WMMA GEMMs, f32 atomics for
// graph scatter/pool, fused bias+BN+leakyReLU epilogues.
// ---------------------------------------------------------------------------

#define NNODES 100000
#define NEDGES 600000
#define NGRAPH 4096
#define DIM    128
#define HID    1024
#define BN_EPS 1e-5f

typedef __bf16 v16bf __attribute__((ext_vector_type(16)));
typedef float  v8f   __attribute__((ext_vector_type(8)));

// ---------------------------------------------------------------------------
// 1) Embedding gather: h[i,:] = emb[x[i],:]   (one wave per node, float4/lane)
// ---------------------------------------------------------------------------
__global__ void embed_kernel(const int* __restrict__ x,
                             const float* __restrict__ emb,
                             float* __restrict__ h) {
    int t    = blockIdx.x * blockDim.x + threadIdx.x;
    int node = t >> 5;
    int c4   = (t & 31) * 4;
    if (node >= NNODES) return;
    const float4 v = *(const float4*)(emb + (size_t)x[node] * DIM + c4);
    *(float4*)(h + (size_t)node * DIM + c4) = v;
}

// ---------------------------------------------------------------------------
// 2) Edge scatter-add: agg[dst[e],:] += h[src[e],:]   (one wave per edge)
// ---------------------------------------------------------------------------
__global__ void scatter_kernel(const float* __restrict__ h,
                               const int* __restrict__ src,
                               const int* __restrict__ dst,
                               float* __restrict__ agg) {
    int wid  = (blockIdx.x * blockDim.x + threadIdx.x) >> 5;
    int lane = threadIdx.x & 31;
    if (wid >= NEDGES) return;
    int s = src[wid];
    int d = dst[wid];
    const float4 v = *(const float4*)(h + (size_t)s * DIM + lane * 4);
    float* p = agg + (size_t)d * DIM + lane * 4;
    atomicAdd(p + 0, v.x);
    atomicAdd(p + 1, v.y);
    atomicAdd(p + 2, v.z);
    atomicAdd(p + 3, v.w);
}

// ---------------------------------------------------------------------------
// 3) Build concatenated conv weight  Wcat[256,128] = [w_rel ; w_root] (f32)
// ---------------------------------------------------------------------------
__global__ void build_wcat_kernel(const float* __restrict__ wrel,
                                  const float* __restrict__ wroot,
                                  float* __restrict__ wcat) {
    int t = blockIdx.x * blockDim.x + threadIdx.x;
    if (t >= 256 * 128) return;
    int k = t >> 7, o = t & 127;
    wcat[t] = (k < 128) ? wrel[k * 128 + o] : wroot[(k - 128) * 128 + o];
}

// ---------------------------------------------------------------------------
// 4) Pre-swizzle W[K,Ncols] (f32) into CDNA5 WMMA B-fragment order (bf16).
//    B 32x16 bf16 order used here (contiguous k, same as A): col = lane&15, K = kt*32 + (lane>>4)*16 + i.
//    Flat: out[((((cb*(K/32)+kt)*8+ct)*32+lane)*16)+i]
// ---------------------------------------------------------------------------
__global__ void wfrag_kernel(const float* __restrict__ W,
                             __bf16* __restrict__ out, int K, int Ncols) {
    int t = blockIdx.x * blockDim.x + threadIdx.x;
    if (t >= K * Ncols) return;
    int i       = t & 15;
    int lane    = (t >> 4) & 31;
    int ct      = (t >> 9) & 7;
    int ktTotal = K >> 5;
    int rest    = t >> 12;            // kt + ktTotal*cb
    int kt      = rest % ktTotal;
    int cb      = rest / ktTotal;
    int kk      = kt * 32 + (lane >> 4) * 16 + i;
    int col     = cb * 128 + ct * 16 + (lane & 15);
    out[t] = (__bf16)W[(size_t)kk * Ncols + col];
}

// ---------------------------------------------------------------------------
// 5) Generic WMMA GEMM: out[M,Ncols] = epi( [A0|A1][M,K] @ Wfrag + bias )
//    Block = 256 threads (8 waves), tile = 128 rows x 128 cols.
//    A staged as bf16 in LDS (row-major, stride 256); K chunked by 256.
//    Per k-step: preload all 8 B fragments (one clause of 16 b128 loads),
//    then 8 back-to-back WMMAs -> matrix pipe not stalled per-tile on L2.
//    Epilogue: +bias, optional eval-BN, optional leaky-relu(0.01).
// ---------------------------------------------------------------------------
__global__ __launch_bounds__(256)
void wmma_gemm_kernel(const float* __restrict__ A0, const float* __restrict__ A1,
                      int Ksplit, int lda0, int lda1,
                      const __bf16* __restrict__ Wfrag,
                      const float* __restrict__ bias,
                      const float* __restrict__ bnm, const float* __restrict__ bnv,
                      const float* __restrict__ bng, const float* __restrict__ bnb,
                      float* __restrict__ out, int M, int K, int Ncols, int relu) {
    __shared__ __bf16 Alds[128 * 256];   // 64 KB

    const int tid     = threadIdx.x;
    const int wave    = tid >> 5;
    const int lane    = tid & 31;
    const int half    = lane >> 4;
    const int nn      = lane & 15;
    const int rowBase = blockIdx.x * 128;
    const int cb      = blockIdx.y;
    const int ktTotal = K >> 5;

    v8f acc[8];
#pragma unroll
    for (int ct = 0; ct < 8; ++ct)
        acc[ct] = (v8f){0.f, 0.f, 0.f, 0.f, 0.f, 0.f, 0.f, 0.f};

    for (int k0 = 0; k0 < K; k0 += 256) {
        int kc = K - k0; if (kc > 256) kc = 256;
        const int gPerRow = kc >> 2;               // float4 groups per row
        const int groups  = 128 * gPerRow;
        for (int g = tid; g < groups; g += 256) {
            int r    = g / gPerRow;
            int kq   = g - r * gPerRow;
            int k    = k0 + (kq << 2);
            int grow = rowBase + r;
            float4 v = make_float4(0.f, 0.f, 0.f, 0.f);
            if (grow < M) {
                const float* sp = (k < Ksplit)
                    ? (A0 + (size_t)grow * lda0 + k)
                    : (A1 + (size_t)grow * lda1 + (k - Ksplit));
                v = *(const float4*)sp;
            }
            union { __bf16 b[4]; uint2 u; } pk;
            pk.b[0] = (__bf16)v.x; pk.b[1] = (__bf16)v.y;
            pk.b[2] = (__bf16)v.z; pk.b[3] = (__bf16)v.w;
            *(uint2*)(Alds + r * 256 + (kq << 2)) = pk.u;
        }
        __syncthreads();

        const int ksteps = kc >> 5;
        for (int kt = 0; kt < ksteps; ++kt) {
            // ---- preload all 8 B fragments for this k-step (coalesced b128s)
            const size_t ktg = (size_t)cb * ktTotal + (size_t)(k0 >> 5) + kt;
            const __bf16* bbase = Wfrag + ((ktg * 8) * 32 + lane) * 16;
            union { v16bf v; uint4 q[2]; } bfr[8];
#pragma unroll
            for (int ct = 0; ct < 8; ++ct) {
                bfr[ct].q[0] = *(const uint4*)(bbase + ct * (32 * 16));
                bfr[ct].q[1] = *(const uint4*)(bbase + ct * (32 * 16) + 8);
            }

            // ---- A fragment from LDS: 2x ds_load_b128
            const __bf16* ap = Alds + (wave * 16 + nn) * 256 + kt * 32 + half * 16;
            union { v16bf v; uint4 q[2]; } a;
            a.q[0] = *(const uint4*)(ap);
            a.q[1] = *(const uint4*)(ap + 8);

            // ---- 8 back-to-back WMMAs (no per-tile load wait)
#pragma unroll
            for (int ct = 0; ct < 8; ++ct) {
                acc[ct] = __builtin_amdgcn_wmma_f32_16x16x32_bf16(
                    false, a.v, false, bfr[ct].v, (short)0, acc[ct], false, false);
            }
        }
        __syncthreads();
    }

    // ---- epilogue: C/D layout (8 f32/lane): row M = half*8 + r, col = lane&15
#pragma unroll
    for (int ct = 0; ct < 8; ++ct) {
        int col  = cb * 128 + ct * 16 + nn;
        float bb = bias ? bias[col] : 0.f;
        float sm = 0.f, srs = 1.f, sg = 1.f, sb = 0.f;
        if (bnm) {
            sm = bnm[col]; srs = rsqrtf(bnv[col] + BN_EPS);
            sg = bng[col]; sb = bnb[col];
        }
#pragma unroll
        for (int r = 0; r < 8; ++r) {
            int grow = rowBase + wave * 16 + half * 8 + r;
            if (grow < M) {
                float v = acc[ct][r] + bb;
                v = (v - sm) * srs * sg + sb;
                if (relu) v = v > 0.f ? v : 0.01f * v;
                out[(size_t)grow * Ncols + col] = v;
            }
        }
    }
}

// ---------------------------------------------------------------------------
// 6) Mean pool: pooled[b,:] += h[i,:], cnt[b] += 1; then divide.
// ---------------------------------------------------------------------------
__global__ void pool_kernel(const float* __restrict__ h,
                            const int* __restrict__ batch,
                            float* __restrict__ pooled,
                            float* __restrict__ cnt) {
    int wid  = (blockIdx.x * blockDim.x + threadIdx.x) >> 5;
    int lane = threadIdx.x & 31;
    if (wid >= NNODES) return;
    int b = batch[wid];
    const float4 v = *(const float4*)(h + (size_t)wid * DIM + lane * 4);
    float* p = pooled + (size_t)b * DIM + lane * 4;
    atomicAdd(p + 0, v.x);
    atomicAdd(p + 1, v.y);
    atomicAdd(p + 2, v.z);
    atomicAdd(p + 3, v.w);
    if (lane == 0) atomicAdd(cnt + b, 1.0f);
}

__global__ void pool_div_kernel(float* __restrict__ pooled,
                                const float* __restrict__ cnt) {
    int t = blockIdx.x * blockDim.x + threadIdx.x;
    if (t >= NGRAPH * DIM) return;
    pooled[t] /= fmaxf(cnt[t >> 7], 1.0f);
}

// ---------------------------------------------------------------------------
// 7) Final head: out[g] = sum_c lrelu(x2[g,c]) * w3[c] + b3  (one wave/graph)
// ---------------------------------------------------------------------------
__global__ void final_kernel(const float* __restrict__ x2,
                             const float* __restrict__ w3,
                             const float* __restrict__ b3,
                             float* __restrict__ out) {
    int wid  = (blockIdx.x * blockDim.x + threadIdx.x) >> 5;
    int lane = threadIdx.x & 31;
    if (wid >= NGRAPH) return;
    const float* row = x2 + (size_t)wid * DIM;
    float s = 0.f;
#pragma unroll
    for (int j = 0; j < 4; ++j) {
        float v = row[lane + j * 32];
        v = v > 0.f ? v : 0.01f * v;
        s += v * w3[lane + j * 32];
    }
#pragma unroll
    for (int off = 16; off > 0; off >>= 1)
        s += __shfl_xor(s, off, 32);
    if (lane == 0) out[wid] = s + b3[0];
}

// ---------------------------------------------------------------------------
// Host launcher
// ---------------------------------------------------------------------------
extern "C" void kernel_launch(void* const* d_in, const int* in_sizes, int n_in,
                              void* d_out, int out_size, void* d_ws, size_t ws_size,
                              hipStream_t stream) {
    (void)in_sizes; (void)n_in; (void)out_size; (void)ws_size;

    const int*   x      = (const int*)d_in[0];
    const int*   edge   = (const int*)d_in[1];         // [2,E] row-major
    const int*   batch  = (const int*)d_in[2];
    const float* emb    = (const float*)d_in[3];
    const float* w_rel[3]  = {(const float*)d_in[4],  (const float*)d_in[7],  (const float*)d_in[10]};
    const float* b_rel[3]  = {(const float*)d_in[5],  (const float*)d_in[8],  (const float*)d_in[11]};
    const float* w_root[3] = {(const float*)d_in[6],  (const float*)d_in[9],  (const float*)d_in[12]};
    const float* w_lin1 = (const float*)d_in[13];
    const float* b_lin1 = (const float*)d_in[14];
    const float* w_lin2 = (const float*)d_in[15];
    const float* b_lin2 = (const float*)d_in[16];
    const float* w_lin3 = (const float*)d_in[17];
    const float* b_lin3 = (const float*)d_in[18];
    const float* bn1_g = (const float*)d_in[19];
    const float* bn1_b = (const float*)d_in[20];
    const float* bn1_m = (const float*)d_in[21];
    const float* bn1_v = (const float*)d_in[22];
    const float* bn2_g = (const float*)d_in[23];
    const float* bn2_b = (const float*)d_in[24];
    const float* bn2_m = (const float*)d_in[25];
    const float* bn2_v = (const float*)d_in[26];
    float* outp = (float*)d_out;

    // ---- carve workspace (256B aligned slabs)
    size_t off = 0;
    auto carve = [&](size_t bytes) {
        void* p = (char*)d_ws + off;
        off += (bytes + 255) & ~(size_t)255;
        return p;
    };
    float*  hA       = (float*)carve((size_t)NNODES * DIM * 4);
    float*  hB       = (float*)carve((size_t)NNODES * DIM * 4);
    float*  agg      = (float*)carve((size_t)NNODES * DIM * 4);
    float*  pooled   = (float*)carve((size_t)NGRAPH * DIM * 4);
    float*  cnt      = (float*)carve((size_t)NGRAPH * 4);
    float*  z        = (float*)carve((size_t)NGRAPH * HID * 4);
    float*  x2       = (float*)carve((size_t)NGRAPH * DIM * 4);
    float*  wcat_f32 = (float*)carve((size_t)256 * 128 * 4);
    __bf16* wcat_fr  = (__bf16*)carve((size_t)256 * 128 * 2);
    __bf16* w1_fr    = (__bf16*)carve((size_t)128 * 1024 * 2);
    __bf16* w2_fr    = (__bf16*)carve((size_t)1024 * 128 * 2);

    const int src_off = 0, dst_off = NEDGES;

    // 1) embedding
    embed_kernel<<<(NNODES * 32 + 255) / 256, 256, 0, stream>>>(x, emb, hA);

    // 2) pre-swizzle MLP weights once
    wfrag_kernel<<<(128 * 1024 + 255) / 256, 256, 0, stream>>>(w_lin1, w1_fr, 128, 1024);
    wfrag_kernel<<<(1024 * 128 + 255) / 256, 256, 0, stream>>>(w_lin2, w2_fr, 1024, 128);

    // 3) three GraphConv layers (ping-pong hA/hB)
    float* hin  = hA;
    float* hout = hB;
    const int convRowBlocks = (NNODES + 127) / 128;
    for (int l = 0; l < 3; ++l) {
        build_wcat_kernel<<<(256 * 128 + 255) / 256, 256, 0, stream>>>(
            w_rel[l], w_root[l], wcat_f32);
        wfrag_kernel<<<(256 * 128 + 255) / 256, 256, 0, stream>>>(
            wcat_f32, wcat_fr, 256, 128);
        hipMemsetAsync(agg, 0, (size_t)NNODES * DIM * 4, stream);
        scatter_kernel<<<(NEDGES * 32 + 255) / 256, 256, 0, stream>>>(
            hin, edge + src_off, edge + dst_off, agg);
        wmma_gemm_kernel<<<dim3(convRowBlocks, 1), 256, 0, stream>>>(
            agg, hin, /*Ksplit=*/128, /*lda0=*/128, /*lda1=*/128,
            wcat_fr, b_rel[l],
            nullptr, nullptr, nullptr, nullptr,
            hout, NNODES, /*K=*/256, /*Ncols=*/128, /*relu=*/1);
        float* t = hin; hin = hout; hout = t;
    }
    const float* hfin = hin;   // after 3 swaps: hB

    // 4) mean pool
    hipMemsetAsync(pooled, 0, (size_t)NGRAPH * DIM * 4, stream);
    hipMemsetAsync(cnt, 0, (size_t)NGRAPH * 4, stream);
    pool_kernel<<<(NNODES * 32 + 255) / 256, 256, 0, stream>>>(hfin, batch, pooled, cnt);
    pool_div_kernel<<<(NGRAPH * DIM + 255) / 256, 256, 0, stream>>>(pooled, cnt);

    // 5) lin1 + BN1 + lrelu  ->  z[4096,1024]
    wmma_gemm_kernel<<<dim3(NGRAPH / 128, HID / 128), 256, 0, stream>>>(
        pooled, pooled, /*Ksplit=*/128, /*lda0=*/128, /*lda1=*/128,
        w1_fr, b_lin1, bn1_m, bn1_v, bn1_g, bn1_b,
        z, NGRAPH, /*K=*/128, /*Ncols=*/HID, /*relu=*/1);

    // 6) lin2 + BN2 (no relu)  ->  x2[4096,128]
    wmma_gemm_kernel<<<dim3(NGRAPH / 128, 1), 256, 0, stream>>>(
        z, z, /*Ksplit=*/HID, /*lda0=*/HID, /*lda1=*/HID,
        w2_fr, b_lin2, bn2_m, bn2_v, bn2_g, bn2_b,
        x2, NGRAPH, /*K=*/HID, /*Ncols=*/128, /*relu=*/0);

    // 7) head
    final_kernel<<<(NGRAPH * 32 + 255) / 256, 256, 0, stream>>>(x2, w_lin3, b_lin3, outp);
}
